// adaptive_loss_48713519071874
// MI455X (gfx1250) — matrix operand order repaired and checked by reading, not compiled, not run on hardware
//
#include <hip/hip_runtime.h>
#include <stdint.h>

typedef __bf16 v16bf __attribute__((ext_vector_type(16)));
typedef float  v8f   __attribute__((ext_vector_type(8)));

#define N_ROWS   4096
#define D_PROJ_  1024
#define D_EMB    512
#define C0       10000
#define C1       30000
#define NTOK     50000
#define WBF_ROWS 50016   // 50000 weight + 2 cluster + 14 zero pad
#define HEAD_COLS 10016  // 10000 + 2 + 14 pad (pad cols get bias = -inf)
#define TAIL_COLS 20000
#define LSE_PAD  8       // bf16 elems of row padding (keeps 16B alignment)

__device__ __forceinline__ unsigned f2bf(float f) {
  union { float f; unsigned u; } v; v.f = f;
  unsigned r = v.u + 0x7fffu + ((v.u >> 16) & 1u);  // round-to-nearest-even
  return r >> 16;
}
__device__ __forceinline__ float bf2f(unsigned short hw) {
  union { unsigned u; float f; } v; v.u = ((unsigned)hw) << 16; return v.f;
}

// ---------------------------------------------------------------- kernel 0
// One pass: build bf16 weight table (+cluster rows +zero pad), bf16 hidden,
// and bf16 *transposed* projections pbfT[p][n][k] so GEMM B-tiles are
// contiguous along K and can be async-copied to LDS without reshuffling.
__global__ void convert_all(const float* __restrict__ weight,
                            const float* __restrict__ cw,
                            const float* __restrict__ hidden,
                            const float* __restrict__ projs,
                            unsigned short* __restrict__ wbf,
                            unsigned short* __restrict__ hbf16,
                            unsigned short* __restrict__ pbfT) {
  const long R0 = (long)WBF_ROWS * D_EMB;
  const long R1 = (long)N_ROWS * D_PROJ_;
  const long R2 = 3L * D_PROJ_ * D_EMB;
  const long total = R0 + R1 + R2;
  for (long i = (long)blockIdx.x * blockDim.x + threadIdx.x; i < total;
       i += (long)gridDim.x * blockDim.x) {
    if (i < R0) {
      int row = (int)(i / D_EMB), c = (int)(i % D_EMB);
      float f = 0.f;
      if (row < NTOK)          f = weight[(long)row * D_EMB + c];
      else if (row < NTOK + 2) f = cw[(row - NTOK) * D_EMB + c];
      wbf[i] = (unsigned short)f2bf(f);
    } else if (i < R0 + R1) {
      long j = i - R0;
      hbf16[j] = (unsigned short)f2bf(hidden[j]);
    } else {
      long j = i - R0 - R1;                       // j = ((p*1024)+k)*512 + n
      int  p = (int)(j / ((long)D_PROJ_ * D_EMB));
      long r = j % ((long)D_PROJ_ * D_EMB);
      int  k = (int)(r / D_EMB), n = (int)(r % D_EMB);
      pbfT[((long)p * D_EMB + n) * D_PROJ_ + k] = (unsigned short)f2bf(projs[j]);
    }
  }
}

// ---------------------------------------------------------------- kernel 1
// H_p = bf16( hidden @ projs[p] ), all-bf16 WMMA with async B staging.
__global__ __launch_bounds__(256)
void proj_gemm(const unsigned short* __restrict__ hbf16,
               const unsigned short* __restrict__ pbfT,
               unsigned short* __restrict__ hbf) {
  __shared__ __align__(16) unsigned short Bt[3][64][40];  // [buf][n][k], 15.4 KB
  const int p = blockIdx.z;
  unsigned short* Ho = hbf + (long)p * N_ROWS * D_EMB;
  const unsigned short* BT = pbfT + (long)p * D_EMB * D_PROJ_;
  const int tid = threadIdx.x;
  const int wave = tid >> 5, lane = tid & 31;
  const int half = lane >> 4, lm = lane & 15;
  const int rowBase = blockIdx.x * 128 + wave * 16;
  const int n0 = blockIdx.y * 64;

  const unsigned ldsBase   = (unsigned)(unsigned long long)(const void*)&Bt[0][0][0];
  const unsigned bufStride = 64u * 40u * 2u;
  const int nn = tid >> 2, q = tid & 3;           // 64 rows x 4 x 16B = 4KB/tile

  auto issue = [&](int kt) {
    unsigned long long g = (unsigned long long)(const void*)
        (BT + (long)(n0 + nn) * D_PROJ_ + kt * 32) + (unsigned)(q * 16);
    unsigned d = ldsBase + (unsigned)(kt % 3) * bufStride + (unsigned)nn * 80u
               + (unsigned)q * 16u;
    asm volatile("global_load_async_to_lds_b128 %0, %1, off" :: "v"(d), "v"(g) : "memory");
  };
  issue(0);

  v8f c[4] = {};
  const unsigned short* hrow = hbf16 + (long)(rowBase + lm) * D_PROJ_;
  union Frag { v16bf v; unsigned u[8]; };

  for (int kt = 0; kt < D_PROJ_ / 32; ++kt) {
    if (kt + 1 < D_PROJ_ / 32) {
      issue(kt + 1);
      asm volatile("s_wait_asynccnt 1" ::: "memory");
    } else {
      asm volatile("s_wait_asynccnt 0" ::: "memory");
    }
    __syncthreads();  // triple buffer => one barrier per K-tile

    Frag a;                                        // A frag: 2 aligned b128 loads
    {
      const uint4 a0 = *(const uint4*)(hrow + kt * 32 + 16 * half);
      const uint4 a1 = *(const uint4*)(hrow + kt * 32 + 16 * half + 8);
      a.u[0] = a0.x; a.u[1] = a0.y; a.u[2] = a0.z; a.u[3] = a0.w;
      a.u[4] = a1.x; a.u[5] = a1.y; a.u[6] = a1.z; a.u[7] = a1.w;
    }
    const unsigned short* bbuf = &Bt[kt % 3][0][0];
    Frag b0, b1;
    {
      const unsigned* pb = (const unsigned*)(bbuf + lm * 40 + half * 16);
      #pragma unroll
      for (int v = 0; v < 8; ++v) b0.u[v] = pb[v];
    }
    #pragma unroll
    for (int j = 0; j < 4; ++j) {                  // N tiles, ping-pong B frags
      Frag& cur = (j & 1) ? b1 : b0;
      Frag& nxt = (j & 1) ? b0 : b1;
      if (j < 3) {
        const unsigned* pb = (const unsigned*)(bbuf + ((j + 1) * 16 + lm) * 40 + half * 16);
        #pragma unroll
        for (int v = 0; v < 8; ++v) nxt.u[v] = pb[v];
      }
      c[j] = __builtin_amdgcn_wmma_f32_16x16x32_bf16(false, a.v, false, cur.v,
                                                     (short)0, c[j], false, false);
    }
  }
  #pragma unroll
  for (int j = 0; j < 4; ++j)
    #pragma unroll
    for (int r = 0; r < 8; ++r) {
      int row = rowBase + r + 8 * half;
      int col = n0 + j * 16 + lm;
      Ho[(long)row * D_EMB + col] = (unsigned short)f2bf(c[j][r]);
    }
}

// ---------------------------------------------------------------- kernel 2
// Online logsumexp over vocab tiles, fused with WMMA logit GEMM.
// B tiles (16 weight rows x 512 K bf16) streamed global->LDS with gfx1250
// async loads (ASYNCcnt), triple-buffered, one barrier per tile.
__global__ __launch_bounds__(256)
void lse_kernel(const unsigned short* __restrict__ hbf,
                const unsigned short* __restrict__ wbf,
                const float* __restrict__ bias,
                const float* __restrict__ cbias,
                float* __restrict__ lse) {
  __shared__ __align__(16) unsigned short Btile[3][16][D_EMB + LSE_PAD];
  const int seg = blockIdx.z;
  const int tid = threadIdx.x, wave = tid >> 5, lane = tid & 31;
  const int half = lane >> 4, lm = lane & 15;
  const int rb = blockIdx.x * 128 + wave * 16;
  const unsigned short* H = hbf + (long)seg * N_ROWS * D_EMB;

  // Persistent A fragments: 16 rows x full K=512 (16 frags x 8 VGPRs)
  unsigned a[16][8];
  const unsigned short* hrow = H + (long)(rb + lm) * D_EMB;
  #pragma unroll
  for (int f = 0; f < 16; ++f)
    #pragma unroll
    for (int v = 0; v < 8; ++v) {
      int kg = f * 32 + 2 * v + half * 16;
      a[f][v] = *(const unsigned*)(hrow + kg);
    }

  float m[8], s[8];
  #pragma unroll
  for (int r = 0; r < 8; ++r) { m[r] = -INFINITY; s[r] = 0.f; }

  const int ncols   = (seg == 0) ? HEAD_COLS : TAIL_COLS;
  const int ntiles  = ncols >> 4;
  const int colBase = (seg == 0) ? 0 : (seg == 1 ? C0 : C1);
  const int cchunk  = tid >> 4;   // which of the 16 weight rows this thread stages
  const int kchunk  = tid & 15;   // 64B slice along K

  const unsigned ldsBase   = (unsigned)(unsigned long long)(const void*)&Btile[0][0][0];
  const unsigned bufStride = 16u * (D_EMB + LSE_PAD) * 2u;
  const unsigned rowStride = (D_EMB + LSE_PAD) * 2u;

  auto issue_tile = [&](int t) {
    int colg = t * 16 + cchunk;
    long wrow = (seg == 0) ? ((colg < C0) ? (long)colg : (long)(NTOK + (colg - C0)))
                           : (long)(colBase + colg);
    unsigned long long g = (unsigned long long)(const void*)(wbf + wrow * D_EMB + kchunk * 32);
    unsigned d = ldsBase + (unsigned)(t % 3) * bufStride
                         + (unsigned)cchunk * rowStride + (unsigned)kchunk * 64u;
    asm volatile("global_load_async_to_lds_b128 %0, %1, off"           :: "v"(d), "v"(g) : "memory");
    asm volatile("global_load_async_to_lds_b128 %0, %1, off offset:16" :: "v"(d), "v"(g) : "memory");
    asm volatile("global_load_async_to_lds_b128 %0, %1, off offset:32" :: "v"(d), "v"(g) : "memory");
    asm volatile("global_load_async_to_lds_b128 %0, %1, off offset:48" :: "v"(d), "v"(g) : "memory");
  };

  issue_tile(0);

  union Frag { v16bf v; unsigned u[8]; };

  for (int ct = 0; ct < ntiles; ++ct) {
    if (ct + 1 < ntiles) {
      issue_tile(ct + 1);
      asm volatile("s_wait_asynccnt 4" ::: "memory");  // tile ct complete; ct+1 in flight
    } else {
      asm volatile("s_wait_asynccnt 0" ::: "memory");
    }
    __syncthreads();   // triple buffer => single barrier per tile

    const unsigned* brow =
        (const unsigned*)&Btile[ct % 3][lm][0] + half * 8;  // 32B frag base per lane

    v8f c = {};
    Frag b0, b1;
    #pragma unroll
    for (int v = 0; v < 8; ++v) b0.u[v] = brow[v];          // frag f=0
    #pragma unroll
    for (int f = 0; f < 16; ++f) {                          // K loop, ping-pong B frags
      Frag& cur = (f & 1) ? b1 : b0;
      Frag& nxt = (f & 1) ? b0 : b1;
      if (f < 15) {
        const unsigned* pp = brow + (f + 1) * 16;
        #pragma unroll
        for (int v = 0; v < 8; ++v) nxt.u[v] = pp[v];
      }
      union { v16bf v; unsigned u[8]; } av;
      #pragma unroll
      for (int v = 0; v < 8; ++v) av.u[v] = a[f][v];
      c = __builtin_amdgcn_wmma_f32_16x16x32_bf16(false, av.v, false, cur.v,
                                                  (short)0, c, false, false);
    }

    int colL = ct * 16 + lm;                     // this lane's column
    float bval;
    if (seg == 0) {
      if (colL < C0)            bval = bias[colL];
      else if (colL < C0 + 2)   bval = cbias[colL - C0];
      else                      bval = -INFINITY;   // pad columns vanish
    } else {
      bval = bias[colBase + colL];
    }
    #pragma unroll
    for (int r = 0; r < 8; ++r) {                // online logsumexp
      float z  = c[r] + bval;
      float nm = fmaxf(m[r], z);
      s[r] = s[r] * __expf(m[r] - nm) + __expf(z - nm);
      m[r] = nm;
    }
  }

  // merge the 16 column-classes (lanes within each half) per row
  #pragma unroll
  for (int off = 8; off >= 1; off >>= 1) {
    #pragma unroll
    for (int r = 0; r < 8; ++r) {
      float m2 = __shfl_xor(m[r], off, 32);
      float s2 = __shfl_xor(s[r], off, 32);
      float nm = fmaxf(m[r], m2);
      s[r] = s[r] * __expf(m[r] - nm) + s2 * __expf(m2 - nm);
      m[r] = nm;
    }
  }
  if (lm == 0) {
    #pragma unroll
    for (int r = 0; r < 8; ++r)
      lse[(long)seg * N_ROWS + rb + r + 8 * half] = m[r] + __logf(s[r]);
  }
}

// ---------------------------------------------------------------- kernel 3
__global__ __launch_bounds__(256)
void finalize(const unsigned short* __restrict__ hbf,
              const unsigned short* __restrict__ wbf,
              const float* __restrict__ bias,
              const float* __restrict__ cbias,
              const float* __restrict__ lse,
              const int* __restrict__ target,
              float* __restrict__ out) {
  const int warp = threadIdx.x >> 5;
  const int lane = threadIdx.x & 31;
  const int n = blockIdx.x * 8 + warp;
  if (n >= N_ROWS) return;
  const int t = target[n];
  const int seg = (t < C0) ? 0 : (t < C1 ? 1 : 2);

  auto dotrow = [&](int hseg, long wrow) {
    const unsigned short* h = hbf + ((long)hseg * N_ROWS + n) * D_EMB;
    const unsigned short* w = wbf + wrow * D_EMB;
    float acc = 0.f;
    #pragma unroll 4
    for (int e = lane; e < D_EMB; e += 32) acc += bf2f(h[e]) * bf2f(w[e]);
    #pragma unroll
    for (int off = 16; off >= 1; off >>= 1) acc += __shfl_xor(acc, off, 32);
    return acc;
  };

  float nll;
  if (seg == 0) {
    float z = dotrow(0, t) + bias[t];
    nll = -(z - lse[n]);
  } else {
    float zc = dotrow(0, NTOK + (seg - 1)) + cbias[seg - 1];  // head cluster logit
    float zt = dotrow(seg, t) + bias[t];                      // tail target logit
    nll = -((zc - lse[n]) + (zt - lse[(long)seg * N_ROWS + n]));
  }
  if (lane == 0) out[n] = nll;
}

// ---------------------------------------------------------------- launch
extern "C" void kernel_launch(void* const* d_in, const int* in_sizes, int n_in,
                              void* d_out, int out_size, void* d_ws, size_t ws_size,
                              hipStream_t stream) {
  (void)in_sizes; (void)n_in; (void)out_size; (void)ws_size;
  const float* hidden = (const float*)d_in[0];
  const int*   target = (const int*)d_in[1];
  const float* weight = (const float*)d_in[2];
  const float* bias   = (const float*)d_in[3];
  const float* cw     = (const float*)d_in[4];
  const float* cbias  = (const float*)d_in[5];
  const float* projs  = (const float*)d_in[6];

  char* ws = (char*)d_ws;
  unsigned short* hbf = (unsigned short*)ws;                     // 12.6 MB (3 proj outputs)
  ws += 3ull * N_ROWS * D_EMB * 2;
  unsigned short* wbf = (unsigned short*)ws;                     // 51.2 MB (bf16 weights)
  ws += (size_t)WBF_ROWS * D_EMB * 2;
  float* lse = (float*)ws;                                       // 48 KB
  ws += 3ull * N_ROWS * sizeof(float);
  unsigned short* hbf16 = (unsigned short*)ws;                   // 8 MB (bf16 hidden)
  ws += (size_t)N_ROWS * D_PROJ_ * 2;
  unsigned short* pbfT = (unsigned short*)ws;                    // 3 MB (bf16 projs^T)

  hipLaunchKernelGGL(convert_all, dim3(8192), dim3(256), 0, stream,
                     weight, cw, hidden, projs, wbf, hbf16, pbfT);
  hipLaunchKernelGGL(proj_gemm, dim3(N_ROWS / 128, D_EMB / 64, 3), dim3(256), 0, stream,
                     hbf16, pbfT, hbf);
  hipLaunchKernelGGL(lse_kernel, dim3(N_ROWS / 128, 1, 3), dim3(256), 0, stream,
                     hbf, wbf, bias, cbias, lse);
  hipLaunchKernelGGL(finalize, dim3(N_ROWS / 8), dim3(256), 0, stream,
                     hbf, wbf, bias, cbias, lse, target, (float*)d_out);
}
